// MambaEncoder_77352361001406
// MI455X (gfx1250) — matrix operand order repaired and checked by reading, not compiled, not run on hardware
//
#include <hip/hip_runtime.h>
#include <cstdint>
#include <cstddef>

#define D_MODEL 256
#define D_STATE 64
#define D_INNER 512
#define DT_RANK 16
#define NLAYERS 4
#define BATCH   8
#define SEQLEN  2048
#define MROWS   (BATCH * SEQLEN)          /* 16384 */
#define NPROJ   (DT_RANK + 2 * D_STATE)   /* 144   */
#define NPROJ_PAD 192                     /* padded to 12x16 so GEMM runs uniform 64-wide strips */

typedef __attribute__((ext_vector_type(16))) _Float16 v16h;
typedef __attribute__((ext_vector_type(8)))  _Float16 v8h;
typedef __attribute__((ext_vector_type(8)))  float    v8f;

// ---------------------------------------------------------------- reductions
__device__ __forceinline__ float warp_sum(float v) {
#pragma unroll
  for (int o = 16; o > 0; o >>= 1) v += __shfl_xor(v, o, 32);
  return v;
}

// blockDim.x == 256 (8 wave32s)
__device__ __forceinline__ float block_sum_256(float v) {
  __shared__ float sm[8];
  const int lane = threadIdx.x & 31, w = threadIdx.x >> 5;
  v = warp_sum(v);
  if (lane == 0) sm[w] = v;
  __syncthreads();
  float t = (lane < 8) ? sm[lane] : 0.0f;
  t = warp_sum(t);          // every wave reduces the 8 partials -> all threads get total
  __syncthreads();          // sm reusable by next call
  return t;
}

// ---------------------------------------------------------------- embed: h = gelu(rmsnorm(x*W0+b0))
__global__ void embed_kernel(const float* __restrict__ x, const float* __restrict__ W0,
                             const float* __restrict__ b0, const float* __restrict__ rms0,
                             float* __restrict__ res) {
  const int row = blockIdx.x;     // b*L + l
  const int c   = threadIdx.x;    // 0..255
  float xv = x[row];
  if (isnan(xv)) xv = 0.0f;
  else if (isinf(xv)) xv = (xv > 0.0f) ? 3.4028235e38f : -3.4028235e38f;
  float v   = xv * W0[c] + b0[c];
  float ssq = block_sum_256(v * v);
  float inv = rsqrtf(ssq * (1.0f / D_MODEL) + 1e-5f);
  float g   = v * inv * rms0[c];
  float h   = 0.5f * g * (1.0f + erff(g * 0.70710678118654752f)); // exact gelu
  res[(size_t)row * D_MODEL + c] = h;
}

// ---------------------------------------------------------------- rmsnorm -> f16 (GEMM A operand)
__global__ void rmsnorm_f16_kernel(const float* __restrict__ X, const float* __restrict__ scale,
                                   _Float16* __restrict__ Y) {
  const int row = blockIdx.x;
  const int c   = threadIdx.x;
  float v   = X[(size_t)row * D_MODEL + c];
  float ssq = block_sum_256(v * v);
  float inv = rsqrtf(ssq * (1.0f / D_MODEL) + 1e-5f);
  Y[(size_t)row * D_MODEL + c] = (_Float16)(v * inv * scale[c]);
}

// ---------------------------------------------------------------- weight repack: W (KxN f32) -> WT (Npad x K f16), zero-pad n >= N
__global__ void repack_wt_kernel(const float* __restrict__ W, _Float16* __restrict__ WT,
                                 int K, int N, int Npad) {
  int idx = blockIdx.x * blockDim.x + threadIdx.x;   // over Npad*K, WT row-major (n,k)
  if (idx >= Npad * K) return;
  int n = idx / K, k = idx - n * K;
  WT[idx] = (n < N) ? (_Float16)W[(size_t)k * N + n] : (_Float16)0.0f;
}

// ---------------------------------------------------------------- WMMA GEMM: C(MxN f32) (+)= A(MxK f16) * BT(NxK f16)^T
// One wave computes a 16 x (16*NTILES) strip; NTILES accumulator tiles share one A fragment.
// NTILES and ACCUM are compile-time so there are no conditional WMMAs and no accumulator copies.
// Fragment shapes per CDNA5 ISA 7.12.2 (wave32, 16-bit A 16x32, B 32x16, f32 C 16x16); A/B use contiguous k.
template <int NTILES, bool ACCUM>
__global__ void __launch_bounds__(128)
wmma_gemm_kernel(const _Float16* __restrict__ A, const _Float16* __restrict__ BT,
                 float* __restrict__ C, int M, int N, int K) {
  const int lane = threadIdx.x & 31;
  const int wave = threadIdx.x >> 5;
  const int m0 = (blockIdx.y * 4 + wave) * 16;
  const int n0 = blockIdx.x * (16 * NTILES);

  const int arow = m0 + (lane & 15);        // A: lane -> row M
  const int kh   = (lane >> 4) * 8;         // A: lane half -> K sub-offset
  const int bn   = lane & 15;               // B/C: lane -> column N
  const int bkh  = (lane >> 4) * 16;        // B: lane half -> K offset
  const int crow = m0 + 8 * (lane >> 4);    // C: vgpr r -> row crow + r

  v8f acc[NTILES];
#pragma unroll
  for (int t = 0; t < NTILES; ++t) {
    if (ACCUM) {
#pragma unroll
      for (int r = 0; r < 8; ++r)
        acc[t][r] = C[(size_t)(crow + r) * N + n0 + t * 16 + bn];
    } else {
#pragma unroll
      for (int r = 0; r < 8; ++r) acc[t][r] = 0.0f;
    }
  }

  for (int k0 = 0; k0 < K; k0 += 32) {
    // A fragment: 16 contiguous K-halves, K = 2*kh+0..15
    const _Float16* pa = A + (size_t)arow * K + k0;
    const v8h alo = *(const v8h*)(pa + 2 * kh);
    const v8h ahi = *(const v8h*)(pa + 2 * kh + 8);
    const v16h af = __builtin_shufflevector(alo, ahi, 0, 1, 2, 3, 4, 5, 6, 7,
                                            8, 9, 10, 11, 12, 13, 14, 15);
#pragma unroll
    for (int t = 0; t < NTILES; ++t) {
      // B fragment: 16 contiguous K-halves of column (n0+t*16+bn) from transposed weights
      const v16h bf = *(const v16h*)(BT + (size_t)(n0 + t * 16 + bn) * K + k0 + bkh);
      acc[t] = __builtin_amdgcn_wmma_f32_16x16x32_f16(false, af, false, bf,
                                                      (short)0, acc[t], false, false);
    }
  }

#pragma unroll
  for (int t = 0; t < NTILES; ++t) {
#pragma unroll
    for (int r = 0; r < 8; ++r)
      C[(size_t)(crow + r) * N + n0 + t * 16 + bn] = acc[t][r];
  }
}

// ---------------------------------------------------------------- causal depthwise conv (width 4) + SiLU
__global__ void conv_silu_kernel(const float* __restrict__ xz, const float* __restrict__ cw,
                                 const float* __restrict__ cb, float* __restrict__ xc32,
                                 _Float16* __restrict__ xc16) {
  int idx = blockIdx.x * blockDim.x + threadIdx.x;
  if (idx >= MROWS * D_INNER) return;
  int d = idx & (D_INNER - 1);
  int m = idx >> 9;              // D_INNER = 512
  int l = m & (SEQLEN - 1);
  int b = m >> 11;               // SEQLEN = 2048
  float s = cb[d];
#pragma unroll
  for (int j = 0; j < 4; ++j) {
    int ls = l - 3 + j;
    if (ls >= 0)
      s += cw[d * 4 + j] * xz[((size_t)(b * SEQLEN + ls)) * (2 * D_INNER) + d];
  }
  float r = s / (1.0f + expf(-s));   // silu
  xc32[idx] = r;
  xc16[idx] = (_Float16)r;
}

// ---------------------------------------------------------------- dt = softplus(dtr @ W_dt + b_dt)   (K=16: plain VALU)
__global__ void dt_kernel(const float* __restrict__ proj, const float* __restrict__ Wdt,
                          const float* __restrict__ bdt, float* __restrict__ dt) {
  int idx = blockIdx.x * blockDim.x + threadIdx.x;
  if (idx >= MROWS * D_INNER) return;
  int d = idx & (D_INNER - 1);
  int m = idx >> 9;
  float s = bdt[d];
  const float* pr = proj + (size_t)m * NPROJ_PAD;
#pragma unroll
  for (int r = 0; r < DT_RANK; ++r) s += pr[r] * Wdt[r * D_INNER + d];
  dt[idx] = (s > 20.0f) ? s : log1pf(expf(s));
}

// ---------------------------------------------------------------- selective scan, fused output gating
// grid (D_INNER/256, BATCH); each thread owns one channel d with all 64 states in VGPRs.
// B_t / C_t (shared across the 512 channels of a batch) broadcast through LDS.
__global__ void __launch_bounds__(256)
scan_kernel(const float* __restrict__ xc, const float* __restrict__ dtb,
            const float* __restrict__ proj, const float* __restrict__ Alog,
            const float* __restrict__ Dsk, const float* __restrict__ xz,
            _Float16* __restrict__ yg) {
  const int b = blockIdx.y;
  const int d = blockIdx.x * blockDim.x + threadIdx.x;
  __shared__ float sB[D_STATE];
  __shared__ float sC[D_STATE];
  float A2[D_STATE], h[D_STATE];
  const float LOG2E = 1.4426950408889634f;
#pragma unroll
  for (int s = 0; s < D_STATE; ++s) {
    A2[s] = -expf(Alog[(size_t)d * D_STATE + s]) * LOG2E;  // A * log2(e), for exp2
    h[s]  = 0.0f;
  }
  const float dskip = Dsk[d];
  for (int l = 0; l < SEQLEN; ++l) {
    const size_t m = (size_t)b * SEQLEN + l;
    __syncthreads();
    if (threadIdx.x < D_STATE)
      sB[threadIdx.x] = proj[m * NPROJ_PAD + DT_RANK + threadIdx.x];
    else if (threadIdx.x < 2 * D_STATE)
      sC[threadIdx.x - D_STATE] = proj[m * NPROJ_PAD + DT_RANK + D_STATE + (threadIdx.x - D_STATE)];
    __syncthreads();
    const float x_t  = xc [m * D_INNER + d];
    const float dt_t = dtb[m * D_INNER + d];
    const float u    = dt_t * x_t;
    float y = 0.0f;
#pragma unroll
    for (int s = 0; s < D_STATE; ++s) {
      h[s] = h[s] * exp2f(dt_t * A2[s]) + u * sB[s];
      y   += h[s] * sC[s];
    }
    y += x_t * dskip;
    const float z = xz[m * (2 * D_INNER) + D_INNER + d];
    y *= z / (1.0f + expf(-z));                  // * silu(z)
    yg[m * D_INNER + d] = (_Float16)y;
  }
}

// ---------------------------------------------------------------- final layernorm, last token only
__global__ void final_ln_kernel(const float* __restrict__ res, const float* __restrict__ w,
                                const float* __restrict__ bb, float* __restrict__ out) {
  const int b = blockIdx.x;
  const int c = threadIdx.x;
  const float* xr = res + ((size_t)b * SEQLEN + (SEQLEN - 1)) * D_MODEL;
  float v  = xr[c];
  float mu = block_sum_256(v) * (1.0f / D_MODEL);
  float cd = v - mu;
  float var = block_sum_256(cd * cd) * (1.0f / D_MODEL);
  out[b * D_MODEL + c] = cd * rsqrtf(var + 1e-5f) * w[c] + bb[c];
}

// ---------------------------------------------------------------- host orchestration
extern "C" void kernel_launch(void* const* d_in, const int* in_sizes, int n_in,
                              void* d_out, int out_size, void* d_ws, size_t ws_size,
                              hipStream_t stream) {
  (void)in_sizes; (void)n_in; (void)out_size; (void)ws_size;
  const float* x     = (const float*)d_in[0];
  const float* W0    = (const float*)d_in[1];
  const float* b0    = (const float*)d_in[2];
  const float* rms0  = (const float*)d_in[3];
  const float* nsc   = (const float*)d_in[4];
  const float* W_in  = (const float*)d_in[5];
  const float* cw    = (const float*)d_in[6];
  const float* cb    = (const float*)d_in[7];
  const float* W_x   = (const float*)d_in[8];
  const float* W_dt  = (const float*)d_in[9];
  const float* b_dt  = (const float*)d_in[10];
  const float* A_log = (const float*)d_in[11];
  const float* Dsk   = (const float*)d_in[12];
  const float* W_out = (const float*)d_in[13];
  const float* ln_w  = (const float*)d_in[14];
  const float* ln_b  = (const float*)d_in[15];

  uint8_t* p = (uint8_t*)d_ws;
  auto carve = [&](size_t bytes) -> void* {
    void* r = (void*)p;
    p += (bytes + 255) & ~(size_t)255;
    return r;
  };
  float*    res   = (float*)   carve((size_t)MROWS * D_MODEL * 4);      // residual stream
  _Float16* xn16  = (_Float16*)carve((size_t)MROWS * D_MODEL * 2);      // rmsnorm'd, f16
  float*    xz    = (float*)   carve((size_t)MROWS * 2 * D_INNER * 4);  // W_in output
  float*    xc32  = (float*)   carve((size_t)MROWS * D_INNER * 4);      // conv+silu
  _Float16* xc16  = (_Float16*)carve((size_t)MROWS * D_INNER * 2);
  float*    proj  = (float*)   carve((size_t)MROWS * NPROJ_PAD * 4);    // [dtr | B | C | pad]
  float*    dtb   = (float*)   carve((size_t)MROWS * D_INNER * 4);
  _Float16* yg16  = (_Float16*)carve((size_t)MROWS * D_INNER * 2);      // gated scan output
  _Float16* WinT  = (_Float16*)carve((size_t)2 * D_INNER * D_MODEL * 2);
  _Float16* WxT   = (_Float16*)carve((size_t)NPROJ_PAD * D_INNER * 2);
  _Float16* WoutT = (_Float16*)carve((size_t)D_MODEL * D_INNER * 2);

  // embedding + first rmsnorm/gelu
  embed_kernel<<<MROWS, 256, 0, stream>>>(x, W0, b0, rms0, res);

  const int EW = MROWS * D_INNER;                  // elementwise work size
  for (int i = 0; i < NLAYERS; ++i) {
    const float* Wi  = W_in  + (size_t)i * D_MODEL * 2 * D_INNER;
    const float* Wxi = W_x   + (size_t)i * D_INNER * NPROJ;
    const float* Woi = W_out + (size_t)i * D_INNER * D_MODEL;

    // transpose+f16 weight repacks (tiny; L2 resident)
    repack_wt_kernel<<<(2 * D_INNER * D_MODEL + 255) / 256, 256, 0, stream>>>(
        Wi, WinT, D_MODEL, 2 * D_INNER, 2 * D_INNER);
    repack_wt_kernel<<<(NPROJ_PAD * D_INNER + 255) / 256, 256, 0, stream>>>(
        Wxi, WxT, D_INNER, NPROJ, NPROJ_PAD);
    repack_wt_kernel<<<(D_MODEL * D_INNER + 255) / 256, 256, 0, stream>>>(
        Woi, WoutT, D_INNER, D_MODEL, D_MODEL);

    // xn = rmsnorm(res) -> f16
    rmsnorm_f16_kernel<<<MROWS, 256, 0, stream>>>(res, nsc + (size_t)i * D_MODEL, xn16);

    // xz = xn @ W_in         (M=16384, N=1024, K=256)
    wmma_gemm_kernel<4, false><<<dim3((2 * D_INNER) / 64, MROWS / 64), 128, 0, stream>>>(
        xn16, WinT, xz, MROWS, 2 * D_INNER, D_MODEL);

    // xc = silu(causal_conv(xz[:, :512]))
    conv_silu_kernel<<<(EW + 255) / 256, 256, 0, stream>>>(
        xz, cw + (size_t)i * D_INNER * 4, cb + (size_t)i * D_INNER, xc32, xc16);

    // proj = xc @ W_x        (M=16384, N=192 padded, K=512)
    wmma_gemm_kernel<4, false><<<dim3(NPROJ_PAD / 64, MROWS / 64), 128, 0, stream>>>(
        xc16, WxT, proj, MROWS, NPROJ_PAD, D_INNER);

    // dt = softplus(dtr @ W_dt + b_dt)
    dt_kernel<<<(EW + 255) / 256, 256, 0, stream>>>(
        proj, W_dt + (size_t)i * DT_RANK * D_INNER, b_dt + (size_t)i * D_INNER, dtb);

    // sequential selective scan + D-skip + silu(z) gating -> f16
    scan_kernel<<<dim3(D_INNER / 256, BATCH), 256, 0, stream>>>(
        xc32, dtb, proj, A_log + (size_t)i * D_INNER * D_STATE,
        Dsk + (size_t)i * D_INNER, xz, yg16);

    // res += yg @ W_out      (M=16384, N=256, K=512), accumulate into res
    wmma_gemm_kernel<4, true><<<dim3(D_MODEL / 64, MROWS / 64), 128, 0, stream>>>(
        yg16, WoutT, res, MROWS, D_MODEL, D_INNER);
  }

  // layernorm of last token of each batch row
  final_ln_kernel<<<BATCH, 256, 0, stream>>>(res, ln_w, ln_b, (float*)d_out);
}
